// EdgeConv_67997922230594
// MI455X (gfx1250) — matrix operand order repaired and checked by reading, not compiled, not run on hardware
//
#include <hip/hip_runtime.h>

typedef __attribute__((ext_vector_type(16))) _Float16 v16h;
typedef __attribute__((ext_vector_type(8)))  float    v8f;

#define BN_EPS 1e-3f

// ---------------------------------------------------------------------------
// Kernel 1: fold BatchNorm affine transforms into weights/biases.
//   BN(x@W) = x@(W*s) + (b - m*s),  s = g*rsqrt(v+eps)
//   BN(x@W + c) = x@(W*s) + ((c - m)*s + b)
// w1 is stored as f16 (WMMA B operand); biases stay f32 (accumulated in f32).
// ---------------------------------------------------------------------------
__global__ void fold_params(
    const float* __restrict__ w_pos,  // [3,32]
    const float* __restrict__ g0, const float* __restrict__ b0,
    const float* __restrict__ m0, const float* __restrict__ v0,
    const float* __restrict__ w_feat, // [16,32]
    const float* __restrict__ gf, const float* __restrict__ bf,
    const float* __restrict__ mf, const float* __restrict__ vf,
    const float* __restrict__ w1,     // [32,32]
    const float* __restrict__ c1,     // b1 [32]
    const float* __restrict__ g1, const float* __restrict__ b1,
    const float* __restrict__ m1, const float* __restrict__ v1,
    float* __restrict__ wpf,    // [3,32]
    float* __restrict__ bias0,  // [32]
    float* __restrict__ wff,    // [16,32]
    float* __restrict__ biasf,  // [32]
    _Float16* __restrict__ w1f, // [32,32]
    float* __restrict__ bias1)  // [32]
{
    int c = threadIdx.x;
    if (c >= 32) return;
    float s0 = g0[c] * rsqrtf(v0[c] + BN_EPS);
    #pragma unroll
    for (int k = 0; k < 3; ++k) wpf[k * 32 + c] = w_pos[k * 32 + c] * s0;
    bias0[c] = b0[c] - m0[c] * s0;

    float sf = gf[c] * rsqrtf(vf[c] + BN_EPS);
    #pragma unroll
    for (int k = 0; k < 16; ++k) wff[k * 32 + c] = w_feat[k * 32 + c] * sf;
    biasf[c] = bf[c] - mf[c] * sf;

    float s1 = g1[c] * rsqrtf(v1[c] + BN_EPS);
    #pragma unroll
    for (int k = 0; k < 32; ++k)
        w1f[k * 32 + c] = (_Float16)(w1[k * 32 + c] * s1);
    bias1[c] = (c1[c] - m1[c]) * s1 + b1[c];
}

// ---------------------------------------------------------------------------
// Kernel 2: node projection  ref_feat2[n][c] = ref_feat[n][:] @ wff[:,c] + biasf[c]
// 51 MFLOP total -> simple one-thread-per-output; wff/biasf are L2/L0 hot.
// ---------------------------------------------------------------------------
__global__ void __launch_bounds__(256) node_proj(
    const float* __restrict__ ref_feat, const float* __restrict__ wff,
    const float* __restrict__ biasf, float* __restrict__ ref_feat2, int nOut)
{
    int idx = blockIdx.x * blockDim.x + threadIdx.x;
    if (idx >= nOut) return;
    int n = idx >> 5, c = idx & 31;
    const float* fr = ref_feat + (long)n * 16;
    float acc = biasf[c];
    #pragma unroll
    for (int k = 0; k < 16; ++k) acc += fr[k] * wff[k * 32 + c];
    ref_feat2[idx] = acc;
}

// ---------------------------------------------------------------------------
// Kernel 3: zero the output (segment-max identity for non-negative values).
// ---------------------------------------------------------------------------
__global__ void __launch_bounds__(256) zero_out(int* __restrict__ out, int n)
{
    int i = blockIdx.x * blockDim.x + threadIdx.x;
    if (i < n) out[i] = 0;
}

// ---------------------------------------------------------------------------
// Kernel 4: main edge kernel. One wave32 per 16-edge tile:
//   A (16x32 f16) = relu(pos_diff@wpf + bias0 + ref_feat2[e_ref])  [built in regs]
//   D (16x32 f32) = A @ w1f via 2x v_wmma_f32_16x16x32_f16
//   out[e_query]  = segment-max via global int-max atomics on float bits.
// Duplicate tiles (clamped tail) are harmless: max is idempotent.
// ---------------------------------------------------------------------------
__global__ void __launch_bounds__(256) edge_kernel(
    const float4* __restrict__ ref_bxyz,   // [Nref]
    const float4* __restrict__ query_bxyz, // [Nq]
    const int*    __restrict__ e_ref,
    const int*    __restrict__ e_query,
    const float*  __restrict__ wpf,        // [3,32] folded
    const float*  __restrict__ bias0,      // [32]
    const _Float16* __restrict__ w1f,      // [32,32] folded, f16
    const float*  __restrict__ bias1,      // [32]
    const float*  __restrict__ ref_feat2,  // [Nref,32]
    int*          __restrict__ out,        // [Nq,32] as int bits
    int E, int tilesPerWave)
{
    const int lane = threadIdx.x & 31;
    const int m    = lane & 15;   // row within tile (edge), or column for B/D
    const int kh   = lane >> 4;   // which K-half this lane holds

    const int wavesPerBlk = blockDim.x >> 5;
    const int gwave = blockIdx.x * wavesPerBlk + (threadIdx.x >> 5);
    const int totalWaves = gridDim.x * wavesPerBlk;
    const int tiles = (E + 15) >> 4;

    // --- per-lane loop-invariant constants ---------------------------------
    // A-fragment channel map: element j<8 -> c = kh*8+j ; j>=8 -> c = 16+kh*8+(j-8)
    float wp0[16], wp1[16], wp2[16], bz[16];
    #pragma unroll
    for (int j = 0; j < 16; ++j) {
        int c = (j < 8) ? (kh * 8 + j) : (16 + kh * 8 + (j - 8));
        wp0[j] = wpf[0 * 32 + c];
        wp1[j] = wpf[1 * 32 + c];
        wp2[j] = wpf[2 * 32 + c];
        bz[j]  = bias0[c];
    }
    // B fragments: lane holds column n = m, elements j<8 -> K = kh*8+j ; j>=8 -> K = 16+kh*8+(j-8)
    v16h bfrag0, bfrag1;
    #pragma unroll
    for (int j = 0; j < 16; ++j) {
        int k = (j < 8) ? (kh * 8 + j) : (16 + kh * 8 + (j - 8));
        bfrag0[j] = w1f[k * 32 + m];
        bfrag1[j] = w1f[k * 32 + m + 16];
    }
    const float bias1n0 = bias1[m];
    const float bias1n1 = bias1[m + 16];

    for (int it = 0; it < tilesPerWave; ++it) {
        int tile = gwave + it * totalWaves;
        if (tile >= tiles) tile = tiles - 1;      // safe: max() idempotent
        long e = (long)tile * 16 + m;
        if (e >= E) e = E - 1;

        const int r = e_ref[e];
        const int q = e_query[e];
        const float4 rb = ref_bxyz[r];
        const float4 qb = query_bxyz[q];
        const float d0 = rb.y - qb.y;
        const float d1 = rb.z - qb.z;
        const float d2 = rb.w - qb.w;

        // gather this lane's 16 channels of ref_feat2[r] (two 32B runs)
        const float* fr = ref_feat2 + (long)r * 32;
        const float4 f0 = *(const float4*)(fr + kh * 8);
        const float4 f1 = *(const float4*)(fr + kh * 8 + 4);
        const float4 f2 = *(const float4*)(fr + 16 + kh * 8);
        const float4 f3 = *(const float4*)(fr + 16 + kh * 8 + 4);
        const float fv[16] = { f0.x, f0.y, f0.z, f0.w, f1.x, f1.y, f1.z, f1.w,
                               f2.x, f2.y, f2.z, f2.w, f3.x, f3.y, f3.z, f3.w };

        // edge_feat = relu(pos_feat + ref_feat2[e_ref]) -> f16 A-fragment
        v16h a;
        #pragma unroll
        for (int j = 0; j < 16; ++j) {
            float v = d0 * wp0[j] + d1 * wp1[j] + d2 * wp2[j] + bz[j] + fv[j];
            v = v > 0.f ? v : 0.f;
            a[j] = (_Float16)v;
        }

        // D = A @ w1f : two 16x16x32 WMMAs cover output columns 0..15, 16..31
        v8f acc0 = {}, acc1 = {};
        acc0 = __builtin_amdgcn_wmma_f32_16x16x32_f16(
            false, a, false, bfrag0, (short)0, acc0, false, false);
        acc1 = __builtin_amdgcn_wmma_f32_16x16x32_f16(
            false, a, false, bfrag1, (short)0, acc1, false, false);

        // D layout: VGPR i -> row (lane<16 ? i : i+8), col = lane&15.
        // bias + relu, then scatter-max (int max == float max for v >= 0).
        #pragma unroll
        for (int i = 0; i < 8; ++i) {
            const int row = (lane < 16) ? i : (i + 8);
            const int qr  = __shfl(q, row, 32);   // e_query of edge `row`
            float v0 = acc0[i] + bias1n0; v0 = v0 > 0.f ? v0 : 0.f;
            float v1 = acc1[i] + bias1n1; v1 = v1 > 0.f ? v1 : 0.f;
            int* dst = out + (long)qr * 32 + m;
            atomicMax(dst,      __float_as_int(v0));
            atomicMax(dst + 16, __float_as_int(v1));
        }
    }
}

// ---------------------------------------------------------------------------
extern "C" void kernel_launch(void* const* d_in, const int* in_sizes, int n_in,
                              void* d_out, int out_size, void* d_ws, size_t ws_size,
                              hipStream_t stream)
{
    const float* ref_bxyz   = (const float*)d_in[0];
    const float* ref_feat   = (const float*)d_in[1];
    const float* query_bxyz = (const float*)d_in[2];
    const int*   e_ref      = (const int*)d_in[3];
    const int*   e_query    = (const int*)d_in[4];
    const float* w_pos = (const float*)d_in[5];
    const float* bn0_g = (const float*)d_in[6];
    const float* bn0_b = (const float*)d_in[7];
    const float* bn0_m = (const float*)d_in[8];
    const float* bn0_v = (const float*)d_in[9];
    const float* w_feat = (const float*)d_in[10];
    const float* bnf_g = (const float*)d_in[11];
    const float* bnf_b = (const float*)d_in[12];
    const float* bnf_m = (const float*)d_in[13];
    const float* bnf_v = (const float*)d_in[14];
    const float* w1    = (const float*)d_in[15];
    const float* b1    = (const float*)d_in[16];
    const float* bn1_g = (const float*)d_in[17];
    const float* bn1_b = (const float*)d_in[18];
    const float* bn1_m = (const float*)d_in[19];
    const float* bn1_v = (const float*)d_in[20];

    const int Nref = in_sizes[0] / 4;
    const int E    = in_sizes[3];

    // workspace layout
    char* ws = (char*)d_ws;
    float* ref_feat2 = (float*)ws;
    size_t off = ((size_t)Nref * 32 * sizeof(float) + 255) & ~(size_t)255;
    float* wpf   = (float*)(ws + off); off += 96  * sizeof(float);
    float* bias0 = (float*)(ws + off); off += 32  * sizeof(float);
    float* wff   = (float*)(ws + off); off += 512 * sizeof(float);
    float* biasf = (float*)(ws + off); off += 32  * sizeof(float);
    float* bias1 = (float*)(ws + off); off += 32  * sizeof(float);
    _Float16* w1f = (_Float16*)(ws + off); off += 1024 * sizeof(_Float16);

    fold_params<<<1, 32, 0, stream>>>(
        w_pos, bn0_g, bn0_b, bn0_m, bn0_v,
        w_feat, bnf_g, bnf_b, bnf_m, bnf_v,
        w1, b1, bn1_g, bn1_b, bn1_m, bn1_v,
        wpf, bias0, wff, biasf, w1f, bias1);

    const int nNode = Nref * 32;
    node_proj<<<(nNode + 255) / 256, 256, 0, stream>>>(
        ref_feat, wff, biasf, ref_feat2, nNode);

    zero_out<<<(out_size + 255) / 256, 256, 0, stream>>>((int*)d_out, out_size);

    const int tiles = (E + 15) >> 4;
    const int blocks = 2048, wavesPerBlk = 8;          // 256 threads/block
    const int totalWaves = blocks * wavesPerBlk;
    const int tilesPerWave = (tiles + totalWaves - 1) / totalWaves;
    edge_kernel<<<blocks, 256, 0, stream>>>(
        (const float4*)ref_bxyz, (const float4*)query_bxyz,
        e_ref, e_query, wpf, bias0, w1f, bias1, ref_feat2,
        (int*)d_out, E, tilesPerWave);
}
